// MultiHeadSelfAttention_85796266705046
// MI455X (gfx1250) — matrix operand order repaired and checked by reading, not compiled, not run on hardware
//
#include <hip/hip_runtime.h>
#include <math.h>

// ---------------------------------------------------------------------------
// Problem constants (fixed by the reference)
// ---------------------------------------------------------------------------
#define Bc   2
#define Lc   1024
#define DIMc 1024
#define Hc   16
#define DHc  64
#define HNc  13           // normal heads; triplet heads are 13,14,15
#define SCALEc 0.125f     // 1/sqrt(64)

typedef __attribute__((ext_vector_type(16))) __bf16 v16bf;
typedef __attribute__((ext_vector_type(8)))  float  v8f;

union FragBF { v16bf v; uint4 u[2]; };

// hardware bf16 convert (backend emits native v_cvt; pairs fuse to pk form)
static __device__ __forceinline__ unsigned short f2bf(float f) {
  return __builtin_bit_cast(unsigned short, static_cast<__bf16>(f));
}
static __device__ __forceinline__ float bf2f(unsigned short u) {
  return __builtin_bit_cast(float, ((unsigned)u) << 16);
}

// ---------------------------------------------------------------------------
// Tiled bf16-WMMA GEMM:  C[M,N] = A[M,K] @ W[N,K]^T + bias[N]
//   a_bf16   : A operand is bf16 (raw copy to LDS) vs f32 (convert on stage)
//   out_mode : 0 = f32 row-major C;  1 = bf16 scatter to (B,H,L,DH) layout
// Block = 256 threads (8 waves), tile 128x128, K-step 32, double-buffered LDS.
// Each wave owns a 16(M)x128(N) strip -> 8 back-to-back WMMAs per K-step.
// ---------------------------------------------------------------------------
__global__ __launch_bounds__(256)
void gemm_bf16_wmma(const void* __restrict__ Ap, int a_bf16,
                    const float* __restrict__ W, const float* __restrict__ bias,
                    void* __restrict__ Cp, int M, int N, int K, int out_mode) {
  __shared__ __align__(16) unsigned short sA[2][128][32];
  __shared__ __align__(16) unsigned short sB[2][128][32];

  const int tid  = threadIdx.x;
  const int wave = tid >> 5;
  const int lane = tid & 31;
  const int mtile = blockIdx.y * 128;
  const int ntile = blockIdx.x * 128;

  auto stageA = [&](int buf, int k0) {
    if (a_bf16) {
      const unsigned short* A = (const unsigned short*)Ap;
#pragma unroll
      for (int j = 0; j < 2; ++j) {
        int idx = tid + j * 256;          // 512 x 16B slots
        int row = idx >> 2;
        int c8  = (idx & 3) << 3;
        *reinterpret_cast<uint4*>(&sA[buf][row][c8]) =
            *reinterpret_cast<const uint4*>(&A[(size_t)(mtile + row) * K + k0 + c8]);
      }
    } else {
      const float* A = (const float*)Ap;
#pragma unroll
      for (int j = 0; j < 4; ++j) {
        int idx = tid + j * 256;
        int row = idx >> 3;
        int kk  = (idx & 7) << 2;
        float4 f = *reinterpret_cast<const float4*>(&A[(size_t)(mtile + row) * K + k0 + kk]);
        sA[buf][row][kk + 0] = f2bf(f.x); sA[buf][row][kk + 1] = f2bf(f.y);
        sA[buf][row][kk + 2] = f2bf(f.z); sA[buf][row][kk + 3] = f2bf(f.w);
      }
    }
  };
  auto stageB = [&](int buf, int k0) {
#pragma unroll
    for (int j = 0; j < 4; ++j) {
      int idx = tid + j * 256;
      int row = idx >> 3;
      int kk  = (idx & 7) << 2;
      float4 g = *reinterpret_cast<const float4*>(&W[(size_t)(ntile + row) * K + k0 + kk]);
      sB[buf][row][kk + 0] = f2bf(g.x); sB[buf][row][kk + 1] = f2bf(g.y);
      sB[buf][row][kk + 2] = f2bf(g.z); sB[buf][row][kk + 3] = f2bf(g.w);
    }
  };

  v8f acc[8] = {};
  const int nkt = K >> 5;

  stageA(0, 0);
  stageB(0, 0);
  __syncthreads();

  for (int kt = 0; kt < nkt; ++kt) {
    const int cur = kt & 1;
    if (kt + 1 < nkt) {                   // overlap next-tile staging with WMMAs
      stageA(cur ^ 1, (kt + 1) << 5);
      stageB(cur ^ 1, (kt + 1) << 5);
    }

    // A fragment: lane<16 -> K{0..7,16..23}, lane>=16 -> K{8..15,24..31}
    FragBF a;
    const int arow = (wave << 4) + (lane & 15);
    const int ka   = (lane >> 4) << 3;
    a.u[0] = *reinterpret_cast<const uint4*>(&sA[cur][arow][ka]);
    a.u[1] = *reinterpret_cast<const uint4*>(&sA[cur][arow][16 + ka]);

    // preload all 8 B fragments, then issue 8 WMMAs back-to-back
    FragBF b[8];
    const int kb = (lane >> 4) << 3;
#pragma unroll
    for (int nn = 0; nn < 8; ++nn) {
      const int bcol = (nn << 4) + (lane & 15);
      b[nn].u[0] = *reinterpret_cast<const uint4*>(&sB[cur][bcol][kb]);
      b[nn].u[1] = *reinterpret_cast<const uint4*>(&sB[cur][bcol][kb + 16]);
    }
#pragma unroll
    for (int nn = 0; nn < 8; ++nn)
      acc[nn] = __builtin_amdgcn_wmma_f32_16x16x32_bf16(false, a.v, false, b[nn].v,
                                                        (short)0, acc[nn], false, false);
    __syncthreads();
  }

#pragma unroll
  for (int nn = 0; nn < 8; ++nn) {
#pragma unroll
    for (int r = 0; r < 8; ++r) {
      int mrow = mtile + (wave << 4) + r + ((lane >> 4) << 3);
      int ncol = ntile + (nn << 4) + (lane & 15);
      float val = acc[nn][r] + bias[ncol];
      if (out_mode == 0) {
        ((float*)Cp)[(size_t)mrow * N + ncol] = val;
      } else {                                      // bf16 (B,H,L,DH) scatter
        int bb = mrow >> 10, l = mrow & (Lc - 1);
        int h  = ncol >> 6,  d = ncol & (DHc - 1);
        ((unsigned short*)Cp)[(((size_t)(bb * Hc + h) << 10) + l) * DHc + d] = f2bf(val);
      }
    }
  }
}

// ---------------------------------------------------------------------------
// Flash attention over the HN normal heads.  Q/K/V already bf16 head-layout.
// K tiles staged global->LDS with CDNA5 async-LDS loads (ASYNCcnt tracked);
// V staged transposed; online softmax; O += P@V via LDS layout round-trip.
// ---------------------------------------------------------------------------
__global__ __launch_bounds__(128)
void attn_wmma(const unsigned short* __restrict__ Q, const unsigned short* __restrict__ K,
               const unsigned short* __restrict__ V, const int* __restrict__ mask,
               unsigned short* __restrict__ ctx) {
  __shared__ __align__(16) unsigned short sK[32][64];      // [key][d]
  __shared__ __align__(16) unsigned short sV[64][32];      // [d][key] (transposed)
  __shared__ __align__(16) unsigned short sP[4][16][32];   // per-wave P scratch
  __shared__ int smask[32];

  const int tid  = threadIdx.x;
  const int wave = tid >> 5;
  const int lane = tid & 31;
  const int b = blockIdx.y / HNc;
  const int h = blockIdx.y % HNc;
  const int qbase = blockIdx.x * 64 + wave * 16;

  const unsigned short* Qh = Q + (size_t)(b * Hc + h) * Lc * DHc;
  const unsigned short* Kh = K + (size_t)(b * Hc + h) * Lc * DHc;
  const unsigned short* Vh = V + (size_t)(b * Hc + h) * Lc * DHc;

  // Q fragments: direct 16B loads, already bf16 in row-major [l][d]
  FragBF qf[2];
  {
    const unsigned short* qp = Qh + (size_t)(qbase + (lane & 15)) * DHc;
#pragma unroll
    for (int c = 0; c < 2; ++c)
#pragma unroll
      for (int half = 0; half < 2; ++half) {
        int dd = c * 32 + half * 16 + ((lane >> 4) << 3);
        qf[c].u[half] = *reinterpret_cast<const uint4*>(qp + dd);
      }
  }

  float mrow[8], lrow[8];
  v8f oacc[4] = {};
#pragma unroll
  for (int r = 0; r < 8; ++r) { mrow[r] = -INFINITY; lrow[r] = 0.f; }

  for (int kt = 0; kt < Lc / 32; ++kt) {
    const int key0 = kt * 32;

    // K tile: async global->LDS copy (no VGPR round-trip, ASYNCcnt tracked)
#pragma unroll
    for (int j = 0; j < 2; ++j) {
      int idx = tid + j * 128;            // 256 x 16B slots
      int key = idx >> 3;
      int d8  = (idx & 7) << 3;
      unsigned       dst = (unsigned)(size_t)&sK[key][d8];
      unsigned long long src =
          (unsigned long long)(size_t)&Kh[(size_t)(key0 + key) * DHc + d8];
      asm volatile("global_load_async_to_lds_b128 %0, %1, off"
                   :: "v"(dst), "v"(src) : "memory");
    }
    // V tile: vector load + transposed LDS scatter
#pragma unroll
    for (int j = 0; j < 2; ++j) {
      int idx = tid + j * 128;
      int key = idx >> 3;
      int d8  = (idx & 7) << 3;
      uint4 gv = *reinterpret_cast<const uint4*>(&Vh[(size_t)(key0 + key) * DHc + d8]);
      const unsigned short* pv = reinterpret_cast<const unsigned short*>(&gv);
#pragma unroll
      for (int i = 0; i < 8; ++i) sV[d8 + i][key] = pv[i];
    }
    if (tid < 32) smask[tid] = mask[b * Lc + key0 + tid];
    asm volatile("s_wait_asynccnt 0" ::: "memory");
    __syncthreads();

    // S = Q @ K^T : two 16x16 key sub-tiles, K-dim = 64 (preload frags)
    FragBF bk[2][2];
#pragma unroll
    for (int t2 = 0; t2 < 2; ++t2)
#pragma unroll
      for (int c = 0; c < 2; ++c) {
        int kcol = t2 * 16 + (lane & 15);
        int kd   = c * 32 + ((lane >> 4) << 3);
        bk[t2][c].u[0] = *reinterpret_cast<const uint4*>(&sK[kcol][kd]);
        bk[t2][c].u[1] = *reinterpret_cast<const uint4*>(&sK[kcol][kd + 16]);
      }
    v8f sacc[2] = {};
#pragma unroll
    for (int t2 = 0; t2 < 2; ++t2)
#pragma unroll
      for (int c = 0; c < 2; ++c)
        sacc[t2] = __builtin_amdgcn_wmma_f32_16x16x32_bf16(false, qf[c].v, false, bk[t2][c].v,
                                                           (short)0, sacc[t2], false, false);

    const bool ok0 = smask[lane & 15] != 0;
    const bool ok1 = smask[16 + (lane & 15)] != 0;

    // online softmax per query row (row r for lanes<16, r+8 for lanes>=16)
#pragma unroll
    for (int r = 0; r < 8; ++r) {
      float s0 = ok0 ? sacc[0][r] * SCALEc : -INFINITY;
      float s1 = ok1 ? sacc[1][r] * SCALEc : -INFINITY;
      float x = fmaxf(s0, s1);
#pragma unroll
      for (int msk = 8; msk >= 1; msk >>= 1) x = fmaxf(x, __shfl_xor(x, msk, 16));
      float mnew = fmaxf(mrow[r], x);
      float corr = __expf(mrow[r] - mnew);
      float p0 = __expf(s0 - mnew);
      float p1 = __expf(s1 - mnew);
      float ps = p0 + p1;
#pragma unroll
      for (int msk = 8; msk >= 1; msk >>= 1) ps += __shfl_xor(ps, msk, 16);
      lrow[r] = lrow[r] * corr + ps;
      mrow[r] = mnew;
#pragma unroll
      for (int nn = 0; nn < 4; ++nn) oacc[nn][r] *= corr;
      int prow = r + ((lane >> 4) << 3);
      sP[wave][prow][lane & 15]        = f2bf(p0);
      sP[wave][prow][16 + (lane & 15)] = f2bf(p1);
    }

    // C-layout -> A-fragment layout via per-wave LDS round-trip
#if __has_builtin(__builtin_amdgcn_s_wait_dscnt)
    __builtin_amdgcn_s_wait_dscnt(0);
#else
    asm volatile("s_wait_dscnt 0" ::: "memory");
#endif
    FragBF pa;
    {
      int prow = lane & 15;
      int pk   = (lane >> 4) << 3;
      pa.u[0] = *reinterpret_cast<const uint4*>(&sP[wave][prow][pk]);
      pa.u[1] = *reinterpret_cast<const uint4*>(&sP[wave][prow][16 + pk]);
    }
    // O += P @ V  (K-dim = 32 keys, 4 d-subtiles, frags preloaded)
    FragBF bv[4];
    const int kk = (lane >> 4) << 3;
#pragma unroll
    for (int nn = 0; nn < 4; ++nn) {
      int dcol = nn * 16 + (lane & 15);
      bv[nn].u[0] = *reinterpret_cast<const uint4*>(&sV[dcol][kk]);
      bv[nn].u[1] = *reinterpret_cast<const uint4*>(&sV[dcol][kk + 16]);
    }
#pragma unroll
    for (int nn = 0; nn < 4; ++nn)
      oacc[nn] = __builtin_amdgcn_wmma_f32_16x16x32_bf16(false, pa.v, false, bv[nn].v,
                                                         (short)0, oacc[nn], false, false);
    __syncthreads();
  }

#pragma unroll
  for (int r = 0; r < 8; ++r) {
    float inv = 1.0f / lrow[r];
    int qrow = qbase + r + ((lane >> 4) << 3);
#pragma unroll
    for (int nn = 0; nn < 4; ++nn) {
      int d = nn * 16 + (lane & 15);
      ctx[((size_t)b * Lc + qrow) * DIMc + h * DHc + d] = f2bf(oacc[nn][r] * inv);
    }
  }
}

// ---------------------------------------------------------------------------
// Triplet prep: Xc[b,d,j,:] = cross(ktr[j+1], ktr[j]); Vt = packed vtr.
// ---------------------------------------------------------------------------
__global__ void triplet_prep(const unsigned short* __restrict__ K,
                             const unsigned short* __restrict__ V,
                             float* __restrict__ Xc, float* __restrict__ Vt) {
  int idx = blockIdx.x * blockDim.x + threadIdx.x;   // (b*64 + d)*1024 + j
  if (idx >= Bc * DHc * Lc) return;
  int j = idx & (Lc - 1);
  int d = (idx >> 10) & 63;
  int b = idx >> 16;
  int j2 = (j + 1) & (Lc - 1);
#define KAT(t, jj) bf2f(K[(((size_t)(b * Hc + HNc + (t)) << 10) + (jj)) * DHc + d])
#define VAT(t)     bf2f(V[(((size_t)(b * Hc + HNc + (t)) << 10) + j) * DHc + d])
  float u0 = KAT(0, j2), u1 = KAT(1, j2), u2 = KAT(2, j2);
  float v0 = KAT(0, j),  v1 = KAT(1, j),  v2 = KAT(2, j);
  size_t o = (size_t)idx * 4;
  Xc[o + 0] = u1 * v2 - u2 * v1;
  Xc[o + 1] = u2 * v0 - u0 * v2;
  Xc[o + 2] = u0 * v1 - u1 * v0;
  Xc[o + 3] = 0.f;
  Vt[o + 0] = VAT(0); Vt[o + 1] = VAT(1); Vt[o + 2] = VAT(2); Vt[o + 3] = 0.f;
#undef KAT
#undef VAT
}

// ---------------------------------------------------------------------------
// Triplet attention: one wave per (b,d,l) row; streaming online softmax,
// butterfly merge; never materializes the 128x(1024x1024) score matrices.
// ---------------------------------------------------------------------------
__global__ __launch_bounds__(256)
void triplet_attn(const unsigned short* __restrict__ Q, const float* __restrict__ Xc,
                  const float* __restrict__ Vt, unsigned short* __restrict__ ctx) {
  int w    = blockIdx.x * 8 + (threadIdx.x >> 5);
  int lane = threadIdx.x & 31;
  int l = w & (Lc - 1);
  int d = (w >> 10) & 63;
  int b = w >> 16;

  float q0 = bf2f(Q[(((size_t)(b * Hc + HNc + 0) << 10) + l) * DHc + d]);
  float q1 = bf2f(Q[(((size_t)(b * Hc + HNc + 1) << 10) + l) * DHc + d]);
  float q2 = bf2f(Q[(((size_t)(b * Hc + HNc + 2) << 10) + l) * DHc + d]);

  const float4* xc = reinterpret_cast<const float4*>(Xc) + ((size_t)(b * DHc + d) << 10);
  const float4* vt = reinterpret_cast<const float4*>(Vt) + ((size_t)(b * DHc + d) << 10);

  float m = -INFINITY, sum = 0.f, a0 = 0.f, a1 = 0.f, a2 = 0.f;
  for (int j = lane; j < Lc; j += 32) {
    float4 c = xc[j];
    float s = fabsf(q0 * c.x + q1 * c.y + q2 * c.z) * SCALEc;
    float4 v = vt[j];
    float mn = fmaxf(m, s);
    float corr = __expf(m - mn);
    float e = __expf(s - mn);
    sum = sum * corr + e;
    a0 = a0 * corr + e * v.x;
    a1 = a1 * corr + e * v.y;
    a2 = a2 * corr + e * v.z;
    m = mn;
  }
#pragma unroll
  for (int msk = 16; msk >= 1; msk >>= 1) {
    float om = __shfl_xor(m, msk, 32);
    float os = __shfl_xor(sum, msk, 32);
    float o0 = __shfl_xor(a0, msk, 32);
    float o1 = __shfl_xor(a1, msk, 32);
    float o2 = __shfl_xor(a2, msk, 32);
    float mn = fmaxf(m, om);
    float e1 = __expf(m - mn), e2 = __expf(om - mn);
    sum = sum * e1 + os * e2;
    a0 = a0 * e1 + o0 * e2;
    a1 = a1 * e1 + o1 * e2;
    a2 = a2 * e1 + o2 * e2;
    m = mn;
  }
  if (lane == 0) {
    float inv = 1.0f / sum;
    size_t o = ((size_t)b * Lc + l) * DIMc + HNc * DHc + d * 3;
    ctx[o + 0] = f2bf(a0 * inv);
    ctx[o + 1] = f2bf(a1 * inv);
    ctx[o + 2] = f2bf(a2 * inv);
  }
}

// ---------------------------------------------------------------------------
// Launch pipeline
// ---------------------------------------------------------------------------
extern "C" void kernel_launch(void* const* d_in, const int* in_sizes, int n_in,
                              void* d_out, int out_size, void* d_ws, size_t ws_size,
                              hipStream_t stream) {
  (void)in_sizes; (void)n_in; (void)out_size; (void)ws_size;
  const float* query = (const float*)d_in[0];
  const float* key   = (const float*)d_in[1];
  const float* value = (const float*)d_in[2];
  const int*   mask  = (const int*)d_in[3];
  const float* Wq = (const float*)d_in[4];  const float* bq = (const float*)d_in[5];
  const float* Wk = (const float*)d_in[6];  const float* bk = (const float*)d_in[7];
  const float* Wv = (const float*)d_in[8];  const float* bv = (const float*)d_in[9];
  const float* Wo = (const float*)d_in[10]; const float* bo = (const float*)d_in[11];
  float* out = (float*)d_out;

  const size_t qkv = (size_t)Bc * Hc * Lc * DHc;          // 2M elements
  unsigned short* Qb  = (unsigned short*)d_ws;            // bf16 workspaces
  unsigned short* Kb  = Qb + qkv;
  unsigned short* Vb  = Kb + qkv;
  unsigned short* ctx = Vb + qkv;                         // B*L*DIM bf16
  float* Xc = (float*)(ctx + (size_t)Bc * Lc * DIMc);     // B*64*L*4 f32
  float* Vt = Xc + (size_t)Bc * DHc * Lc * 4;

  dim3 gblk(256);
  dim3 ggrid(DIMc / 128, (Bc * Lc) / 128);                // 8 x 16 tiles

  gemm_bf16_wmma<<<ggrid, gblk, 0, stream>>>(query, 0, Wq, bq, Qb, Bc * Lc, DIMc, DIMc, 1);
  gemm_bf16_wmma<<<ggrid, gblk, 0, stream>>>(key,   0, Wk, bk, Kb, Bc * Lc, DIMc, DIMc, 1);
  gemm_bf16_wmma<<<ggrid, gblk, 0, stream>>>(value, 0, Wv, bv, Vb, Bc * Lc, DIMc, DIMc, 1);

  triplet_prep<<<(Bc * DHc * Lc + 255) / 256, 256, 0, stream>>>(Kb, Vb, Xc, Vt);

  attn_wmma<<<dim3(Lc / 64, Bc * HNc), 128, 0, stream>>>(Qb, Kb, Vb, mask, ctx);

  triplet_attn<<<dim3((Bc * DHc * Lc) / 8), 256, 0, stream>>>(Qb, Xc, Vt, ctx);

  gemm_bf16_wmma<<<ggrid, gblk, 0, stream>>>(ctx, 1, Wo, bo, out, Bc * Lc, DIMc, DIMc, 0);
}
